// GptOssMLPLearn_28664611734204
// MI455X (gfx1250) — matrix operand order repaired and checked by reading, not compiled, not run on hardware
//
#include <hip/hip_runtime.h>
#include <hip/hip_bf16.h>

// ---------------------------------------------------------------------------
// GPT-OSS MoE MLP for gfx1250 (MI455X), wave32 + v_wmma_f32_16x16x32_bf16.
// Sparse (top-4) expert evaluation; memory-bound on fp32 weight reads.
// ---------------------------------------------------------------------------

typedef __attribute__((ext_vector_type(16))) __bf16 v16bf;
typedef __attribute__((ext_vector_type(8)))  __bf16 v8bf;
typedef __attribute__((ext_vector_type(4)))  __bf16 v4bf;
typedef __attribute__((ext_vector_type(8)))  float  v8f;

static constexpr int E = 32;
static constexpr int TOPK = 4;
static constexpr int H = 1024;
static constexpr int I = 1024;
static constexpr int T = 2048;           // B*S
static constexpr float ALPHA = 1.702f;
static constexpr float LIMIT = 7.0f;
static constexpr int MTILE = 64;         // tokens per workgroup tile
static constexpr int MAX_TILES = T / MTILE;  // 32
static constexpr int BSTRIDE = 36;       // padded LDS stride (bank-conflict-free)

__device__ __forceinline__ v16bf cat8(v8bf lo, v8bf hi) {
  return __builtin_shufflevector(lo, hi, 0,1,2,3,4,5,6,7,8,9,10,11,12,13,14,15);
}

// ---------------------------------------------------------------------------
// Kernel 1: router logits + top-4 softmax + per-expert token lists.
// One wave32 per token; lane == expert.
// ---------------------------------------------------------------------------
__global__ __launch_bounds__(256) void router_kernel(
    const float* __restrict__ x, const float* __restrict__ rw,
    const float* __restrict__ rb, float* __restrict__ scores,
    int* __restrict__ counts, int* __restrict__ tok, float* __restrict__ wgt) {
  const int wave = threadIdx.x >> 5;
  const int lane = threadIdx.x & 31;
  const int t = blockIdx.x * 8 + wave;

  const float* xr = x + (size_t)t * H;
  const float* wr = rw + (size_t)lane * H;
  float acc = 0.f;
  #pragma unroll 4
  for (int k = 0; k < H; k += 4) {
    float4 xv = *(const float4*)(xr + k);
    float4 wv = *(const float4*)(wr + k);
    acc += xv.x * wv.x + xv.y * wv.y + xv.z * wv.z + xv.w * wv.w;
  }
  acc += rb[lane];

  // top-4 with lower-index tie-break (matches lax.top_k)
  float cur = acc;
  int selk = -1;
  float topv[TOPK];
  #pragma unroll
  for (int k = 0; k < TOPK; ++k) {
    float mv = cur;
    int mi = lane;
    #pragma unroll
    for (int off = 16; off > 0; off >>= 1) {
      float ov = __shfl_xor(mv, off, 32);
      int oi = __shfl_xor(mi, off, 32);
      if (ov > mv || (ov == mv && oi < mi)) { mv = ov; mi = oi; }
    }
    topv[k] = mv;
    if (lane == mi) { cur = -3.402823466e38f; selk = k; }
  }
  float e0 = __expf(topv[0] - topv[0]);
  float e1 = __expf(topv[1] - topv[0]);
  float e2 = __expf(topv[2] - topv[0]);
  float e3 = __expf(topv[3] - topv[0]);
  float inv = 1.0f / (e0 + e1 + e2 + e3);
  float wsel = 0.f;
  if (selk == 0) wsel = e0 * inv;
  else if (selk == 1) wsel = e1 * inv;
  else if (selk == 2) wsel = e2 * inv;
  else if (selk == 3) wsel = e3 * inv;

  scores[(size_t)t * E + lane] = wsel;
  if (selk >= 0) {
    int pos = atomicAdd(&counts[lane], 1);
    tok[lane * T + pos] = t;
    wgt[lane * T + pos] = wsel;
  }
}

// ---------------------------------------------------------------------------
// Kernel 2: exclusive prefix sum over PADDED counts (64-row granules) so each
// expert owns a private padded activation region -> unconditional stores.
// ---------------------------------------------------------------------------
__global__ void scan_kernel(const int* __restrict__ counts, int* __restrict__ offsets) {
  if (threadIdx.x == 0) {
    int s = 0;
    for (int e = 0; e < E; ++e) {
      offsets[e] = s;
      s += ((counts[e] + MTILE - 1) / MTILE) * MTILE;
    }
  }
}

// ---------------------------------------------------------------------------
// Kernel 3: gate/up GEMM + clamp/SiLU epilogue -> bf16 activations in ws.
// Grid: (tile, expert). 256 threads = 8 waves; wave owns 8 col-blocks of 16
// gate cols + 16 up cols (interleaved in memory -> one coalesced 128B row
// segment covers both). Weight blocks are software-pipelined: next K-step's
// global loads issue before this step's WMMAs.
// ---------------------------------------------------------------------------
__global__ __launch_bounds__(256) void gateup_kernel(
    const float* __restrict__ x, const float* __restrict__ gup,
    const float* __restrict__ gup_bias, const int* __restrict__ counts,
    const int* __restrict__ offsets, const int* __restrict__ tok,
    __bf16* __restrict__ act) {
  __shared__ __bf16 xs[MTILE][H];                 // 128 KB (CDNA5 320KB LDS)
  __shared__ __bf16 bstg[8][32 * BSTRIDE];        // 18 KB, per-wave transpose
  __shared__ int tl[MTILE];

  const int e = blockIdx.y;
  const int tile = blockIdx.x;
  const int count = counts[e];
  const int base = tile * MTILE;
  if (base >= count) return;
  const int off = offsets[e];
  const int tid = threadIdx.x;

  if (tid < MTILE) {
    int m = base + tid;
    tl[tid] = (m < count) ? tok[e * T + m] : -1;
  }
  __syncthreads();

  // Stage the 64xH token tile as bf16; one row per pass, fully coalesced.
  for (int m = 0; m < MTILE; ++m) {
    int tt = tl[m];
    int k = tid * 4;
    v4bf v4;
    if (tt >= 0) {
      float4 v = *(const float4*)(x + (size_t)tt * H + k);
      v4[0] = (__bf16)v.x; v4[1] = (__bf16)v.y; v4[2] = (__bf16)v.z; v4[3] = (__bf16)v.w;
    } else {
      v4[0] = v4[1] = v4[2] = v4[3] = (__bf16)0.0f;
    }
    *(v4bf*)&xs[m][k] = v4;
  }
  __syncthreads();

  const int wave = tid >> 5;
  const int lane = tid & 31;
  const int nl = lane & 15;
  const int hi16 = lane >> 4;
  const int khalf = hi16 ? 8 : 0;        // unused; A and B both use rbase
  const int rbase = hi16 ? 16 : 0;       // B-matrix per-lane K offset
  const float* wb = gup + (size_t)e * H * (2 * I);
  __bf16* stg = bstg[wave];

  // 4x4 mini-block assignment for the 32(h) x 32(memcol) fp32 weight block.
  const int r0a = ((lane >> 3) + 0) * 4;        // it = 0
  const int c4a = (lane & 7) * 4;
  const int r0b = ((lane >> 3) + 4) * 4;        // it = 1
  const int c4b = c4a;

  for (int cbi = 0; cbi < 8; ++cbi) {
    const int cb = wave * 8 + cbi;       // 0..63 column blocks of 16
    const int n0 = cb * 16;
    v8f accg[4], accu[4];
    #pragma unroll
    for (int mb = 0; mb < 4; ++mb) {
      accg[mb] = (v8f){0.f,0.f,0.f,0.f,0.f,0.f,0.f,0.f};
      accu[mb] = (v8f){0.f,0.f,0.f,0.f,0.f,0.f,0.f,0.f};
    }

    float ga[4][4], gb2[4][4];           // [row_i][col_cc], two mini-blocks
    // prologue: load K-step 0 weight block into registers
    #pragma unroll
    for (int i = 0; i < 4; ++i) {
      float4 v = *(const float4*)(wb + (size_t)(r0a + i) * (2 * I) + 2 * n0 + c4a);
      ga[i][0] = v.x; ga[i][1] = v.y; ga[i][2] = v.z; ga[i][3] = v.w;
      float4 w = *(const float4*)(wb + (size_t)(r0b + i) * (2 * I) + 2 * n0 + c4b);
      gb2[i][0] = w.x; gb2[i][1] = w.y; gb2[i][2] = w.z; gb2[i][3] = w.w;
    }

    for (int k0 = 0; k0 < H; k0 += 32) {
      // Transposed bf16 store: 4 bf16 along K packed into one ds_store_b64.
      #pragma unroll
      for (int cc = 0; cc < 4; ++cc) {
        v4bf p, q;
        #pragma unroll
        for (int i = 0; i < 4; ++i) { p[i] = (__bf16)ga[i][cc]; q[i] = (__bf16)gb2[i][cc]; }
        *(v4bf*)&stg[(c4a + cc) * BSTRIDE + r0a] = p;
        *(v4bf*)&stg[(c4b + cc) * BSTRIDE + r0b] = q;
      }
      // Prefetch next K-step weight block (hidden behind WMMAs below).
      if (k0 + 32 < H) {
        #pragma unroll
        for (int i = 0; i < 4; ++i) {
          float4 v = *(const float4*)(wb + (size_t)(k0 + 32 + r0a + i) * (2 * I) + 2 * n0 + c4a);
          ga[i][0] = v.x; ga[i][1] = v.y; ga[i][2] = v.z; ga[i][3] = v.w;
          float4 w = *(const float4*)(wb + (size_t)(k0 + 32 + r0b + i) * (2 * I) + 2 * n0 + c4b);
          gb2[i][0] = w.x; gb2[i][1] = w.y; gb2[i][2] = w.z; gb2[i][3] = w.w;
        }
      }
      asm volatile("s_wait_dscnt 0" ::: "memory");  // intra-wave LDS exchange

      // B regs: lane nl = column; K-contiguous 16 values (lanes 16-31: K+16).
      v16bf bg, bu;
      #pragma unroll
      for (int q = 0; q < 4; ++q) {
        v4bf g4 = *(const v4bf*)&stg[(2 * nl) * BSTRIDE + rbase + q * 4];
        v4bf u4 = *(const v4bf*)&stg[(2 * nl + 1) * BSTRIDE + rbase + q * 4];
        #pragma unroll
        for (int z = 0; z < 4; ++z) { bg[q * 4 + z] = g4[z]; bu[q * 4 + z] = u4[z]; }
      }

      #pragma unroll
      for (int mb = 0; mb < 4; ++mb) {
        int m = mb * 16 + nl;
        v8bf alo = *(const v8bf*)&xs[m][k0 + rbase];
        v8bf ahi = *(const v8bf*)&xs[m][k0 + rbase + 8];
        v16bf a = cat8(alo, ahi);
        accg[mb] = __builtin_amdgcn_wmma_f32_16x16x32_bf16(
            false, a, false, bg, (short)0, accg[mb], false, false);
        accu[mb] = __builtin_amdgcn_wmma_f32_16x16x32_bf16(
            false, a, false, bu, (short)0, accu[mb], false, false);
      }
    }

    // Epilogue: bias + clamp + SiLU(alpha) -> bf16 activation buffer.
    // Unconditional: padded rows land in this expert's private padding.
    const int gcol = n0 + nl;
    const float gb = gup_bias[(size_t)e * 2 * I + 2 * gcol];
    const float ub = gup_bias[(size_t)e * 2 * I + 2 * gcol + 1];
    const int mro = hi16 ? 8 : 0;
    #pragma unroll
    for (int mb = 0; mb < 4; ++mb) {
      #pragma unroll
      for (int v = 0; v < 8; ++v) {
        int mloc = mb * 16 + mro + v;
        float g = accg[mb][v] + gb;
        float u = accu[mb][v] + ub;
        g = fminf(g, LIMIT);
        u = fminf(fmaxf(u, -LIMIT), LIMIT);
        float glu = g * (1.0f / (1.0f + __expf(-ALPHA * g)));
        float av = (u + 1.0f) * glu;
        act[(size_t)(off + base + mloc) * I + gcol] = (__bf16)av;
      }
    }
  }
}

// ---------------------------------------------------------------------------
// Kernel 4: down GEMM, scale by routing weight, atomic scatter-add to output.
// Branchless epilogue: invalid rows carry weight 0 -> exact +0.0 atomic.
// ---------------------------------------------------------------------------
__global__ __launch_bounds__(256) void down_kernel(
    const __bf16* __restrict__ act, const float* __restrict__ dw,
    const float* __restrict__ dbias, const int* __restrict__ counts,
    const int* __restrict__ offsets, const int* __restrict__ tok,
    const float* __restrict__ wgt, float* __restrict__ out) {
  __shared__ __bf16 bstg[8][16 * BSTRIDE];
  __shared__ int tl[MTILE];
  __shared__ float wl[MTILE];

  const int e = blockIdx.y;
  const int tile = blockIdx.x;
  const int count = counts[e];
  const int base = tile * MTILE;
  if (base >= count) return;
  const int off = offsets[e];
  const int tid = threadIdx.x;

  if (tid < MTILE) {
    int m = base + tid;
    tl[tid] = (m < count) ? tok[e * T + m] : 0;
    wl[tid] = (m < count) ? wgt[e * T + m] : 0.f;
  }
  __syncthreads();

  const int wave = tid >> 5;
  const int lane = tid & 31;
  const int nl = lane & 15;
  const int hi16 = lane >> 4;
  const int khalf = hi16 ? 8 : 0;
  const int rbase = hi16 ? 16 : 0;
  const int mro = hi16 ? 8 : 0;
  const float* wb = dw + (size_t)e * I * H;
  const __bf16* arow = act + (size_t)(off + base) * I;
  __bf16* stg = bstg[wave];

  // 4x4 mini-block assignment for the 32(i) x 16(h) fp32 weight block.
  const int r0 = (lane >> 2) * 4;
  const int c4 = (lane & 3) * 4;

  for (int nbi = 0; nbi < 8; ++nbi) {
    const int n0 = (wave * 8 + nbi) * 16;
    v8f acc[4];
    #pragma unroll
    for (int mb = 0; mb < 4; ++mb)
      acc[mb] = (v8f){0.f,0.f,0.f,0.f,0.f,0.f,0.f,0.f};

    float gg[4][4];
    #pragma unroll
    for (int i = 0; i < 4; ++i) {
      float4 v = *(const float4*)(wb + (size_t)(r0 + i) * H + n0 + c4);
      gg[i][0] = v.x; gg[i][1] = v.y; gg[i][2] = v.z; gg[i][3] = v.w;
    }

    for (int k0 = 0; k0 < I; k0 += 32) {
      #pragma unroll
      for (int cc = 0; cc < 4; ++cc) {
        v4bf p;
        #pragma unroll
        for (int i = 0; i < 4; ++i) p[i] = (__bf16)gg[i][cc];
        *(v4bf*)&stg[(c4 + cc) * BSTRIDE + r0] = p;
      }
      if (k0 + 32 < I) {
        #pragma unroll
        for (int i = 0; i < 4; ++i) {
          float4 v = *(const float4*)(wb + (size_t)(k0 + 32 + r0 + i) * H + n0 + c4);
          gg[i][0] = v.x; gg[i][1] = v.y; gg[i][2] = v.z; gg[i][3] = v.w;
        }
      }
      asm volatile("s_wait_dscnt 0" ::: "memory");

      v16bf b;
      #pragma unroll
      for (int q = 0; q < 4; ++q) {
        v4bf b4 = *(const v4bf*)&stg[nl * BSTRIDE + rbase + q * 4];
        #pragma unroll
        for (int z = 0; z < 4; ++z) b[q * 4 + z] = b4[z];
      }

      #pragma unroll
      for (int mb = 0; mb < 4; ++mb) {
        int m = mb * 16 + nl;
        v8bf alo = *(const v8bf*)(arow + (size_t)m * I + k0 + rbase);
        v8bf ahi = *(const v8bf*)(arow + (size_t)m * I + k0 + rbase + 8);
        v16bf a = cat8(alo, ahi);
        acc[mb] = __builtin_amdgcn_wmma_f32_16x16x32_bf16(
            false, a, false, b, (short)0, acc[mb], false, false);
      }
    }

    const int h = n0 + nl;
    const float bb = dbias[(size_t)e * H + h];
    #pragma unroll
    for (int mb = 0; mb < 4; ++mb) {
      #pragma unroll
      for (int v = 0; v < 8; ++v) {
        int mloc = mb * 16 + mro + v;
        float val = (acc[mb][v] + bb) * wl[mloc];   // 0 for padded rows
        unsafeAtomicAdd(&out[(size_t)tl[mloc] * H + h], val);
      }
    }
  }
}

// ---------------------------------------------------------------------------
// Host launcher (graph-capture safe: memsetAsync + kernel launches only).
// ---------------------------------------------------------------------------
extern "C" void kernel_launch(void* const* d_in, const int* in_sizes, int n_in,
                              void* d_out, int out_size, void* d_ws, size_t ws_size,
                              hipStream_t stream) {
  const float* x      = (const float*)d_in[0];
  const float* gup    = (const float*)d_in[1];
  const float* gup_b  = (const float*)d_in[2];
  const float* dwn    = (const float*)d_in[3];
  const float* dwn_b  = (const float*)d_in[4];
  const float* rw     = (const float*)d_in[5];
  const float* rb     = (const float*)d_in[6];

  float* out    = (float*)d_out;                 // routed: T*H floats
  float* scores = out + (size_t)T * H;           // then T*E floats

  char* ws = (char*)d_ws;
  int*    counts  = (int*)ws;                    // 32 ints
  int*    offsets = (int*)(ws + 128);            // 32 ints (padded prefix)
  int*    tokl    = (int*)(ws + 256);            // E*T ints
  float*  wgtl    = (float*)(ws + 256 + (size_t)E * T * 4);
  __bf16* actb    = (__bf16*)(ws + (1u << 20));  // <=10240 x 1024 bf16 (~21 MB)

  hipMemsetAsync(out, 0, (size_t)T * H * sizeof(float), stream);
  hipMemsetAsync(counts, 0, E * sizeof(int), stream);

  router_kernel<<<T / 8, 256, 0, stream>>>(x, rw, rb, scores, counts, tokl, wgtl);
  scan_kernel<<<1, 32, 0, stream>>>(counts, offsets);
  gateup_kernel<<<dim3(MAX_TILES, E), 256, 0, stream>>>(
      x, gup, gup_b, counts, offsets, tokl, actb);
  down_kernel<<<dim3(MAX_TILES, E), 256, 0, stream>>>(
      actb, dwn, dwn_b, counts, offsets, tokl, wgtl, out);
}
